// StyleBlock_352187318656
// MI455X (gfx1250) — matrix operand order repaired and checked by reading, not compiled, not run on hardware
//
#include <hip/hip_runtime.h>

// ---------------- CDNA5 vector types ----------------
typedef __attribute__((ext_vector_type(16))) __bf16 v16bf;
typedef __attribute__((ext_vector_type(8)))  __bf16 v8bf;
typedef __attribute__((ext_vector_type(8)))  float  v8f;

// ---------------- problem constants ----------------
#define BATCH 8
#define DHW   32
#define CIN   128
#define FOUT  128
#define WDIM  512
#define KTAPS 27                 // 3*3*3
#define KTOT  (KTAPS * CIN)      // 3456 (GEMM K)
#define KSTEPS (KTOT / 32)       // 108 WMMA k-steps
#define NT    (FOUT / 16)        // 8 N tiles
#define MT    8                  // 8 M tiles -> M = 128 spatial per WG
#define HROWS 4                  // 4 output H rows per WG

// padded bf16 input row: w in {-1..32} -> 34 slots, slots 0 and 33 are zero
#define WPAD  34
#define ROWH  (WPAD * CIN)       // 4352 halfs = 8704 B per (b,d,h) row

// LDS A tile: 3 d-slices x 6 h-rows x ROWH
#define LDS_ROWS  (3 * 6)
#define LDS_ELEMS (LDS_ROWS * ROWH)          // 78336 halfs = 156672 B
#define ROW_CHUNKS (ROWH / 8)                // 544 16-byte chunks per row

// =====================================================================
// Kernel 1: style = w_latent @ (w_style/sqrt(512)) + b_style   [8,128]
//           d[b,f] = rsqrt( rc^2 * sum_{tap,c} wconv^2 * style^2 + eps )
// =====================================================================
__global__ __launch_bounds__(CIN)
void style_demod_kernel(const float* __restrict__ w_latent,
                        const float* __restrict__ w_style,
                        const float* __restrict__ b_style,
                        const float* __restrict__ w_conv,
                        float* __restrict__ style_out,
                        float* __restrict__ demod_out) {
    __shared__ float s_lat[WDIM];
    __shared__ float s_style[CIN];
    const int b = blockIdx.x;
    const int f = threadIdx.x;           // 128 threads
    for (int i = f; i < WDIM; i += CIN) s_lat[i] = w_latent[b * WDIM + i];
    __syncthreads();

    const float rc_lin = 0.04419417382415922f;   // 1/sqrt(512)
    float acc = 0.0f;
    for (int k = 0; k < WDIM; ++k)
        acc += s_lat[k] * w_style[k * CIN + f];  // coalesced over f
    const float st = acc * rc_lin + b_style[f];
    s_style[f] = st;
    style_out[b * CIN + f] = st;
    __syncthreads();

    const float rc2 = 1.0f / 3456.0f;            // (1/sqrt(27*128))^2
    float dacc = 0.0f;
    for (int c = 0; c < CIN; ++c) {
        const float s2 = s_style[c] * s_style[c];
        float wsum = 0.0f;
        for (int t = 0; t < KTAPS; ++t) {
            const float w = w_conv[(t * CIN + c) * FOUT + f];
            wsum += w * w;
        }
        dacc += s2 * wsum;
    }
    demod_out[b * CIN + f] = rsqrtf(rc2 * dacc + 1e-8f);
}

// =====================================================================
// Kernel 2: per-batch style-modulated weights in a contiguous-K bf16 B-matrix
// (32x16) lane layout. bpack[b][s][nt][lane][j] (16 halfs = 32B per lane):
//   lane<16 : column n = nt*16+lane,    K_local = j        (K 0..15)
//   lane>=16: column n = nt*16+lane-16, K_local = 16 + j   (K 16..31)
// K_global = 32*s + K_local; tap = s>>2, c = (s&3)*32 + K_local.
// Folding: conv(x*style, w) == conv(x, w * style[c]) per batch.
// =====================================================================
__global__ __launch_bounds__(32)
void pack_w_kernel(const float* __restrict__ w_conv,
                   const float* __restrict__ style,
                   __bf16* __restrict__ bpack) {
    const int blk  = blockIdx.x;         // BATCH*KSTEPS*NT = 6912 blocks
    const int nt   = blk % NT;
    const int s    = (blk / NT) % KSTEPS;
    const int b    = blk / (NT * KSTEPS);
    const int lane = threadIdx.x;        // one wave32
    const int n     = nt * 16 + (lane & 15);
    const int kbase = (lane < 16) ? 0 : 16;
    const int tap   = s >> 2;
    const int cblk  = s & 3;
    const float rc  = 0.017010345219202145f;     // 1/sqrt(3456)
    __bf16* dst = bpack + ((size_t)blk * 32 + lane) * 16;
    #pragma unroll
    for (int j = 0; j < 16; ++j) {
        const int c = cblk * 32 + kbase + j;
        const float w = rc * w_conv[(tap * CIN + c) * FOUT + n] * style[b * CIN + c];
        dst[j] = (__bf16)w;
    }
}

// =====================================================================
// Kernel 3: convert x -> W-padded bf16 rows xw[b][d][h][34][128]
// (slots w=0 and w=33 zero), so conv staging is a pure 16B-chunk copy.
// One block per (b,d,h) row.
// =====================================================================
__global__ __launch_bounds__(256)
void cvt_pad_kernel(const float* __restrict__ x, __bf16* __restrict__ xw) {
    const int row = blockIdx.x;                       // b*1024 + d*32 + h
    const float* xrow = x + (size_t)row * (DHW * CIN);
    __bf16* orow = xw + (size_t)row * ROWH;
    for (int idx = threadIdx.x; idx < ROW_CHUNKS; idx += 256) {
        const int w  = idx >> 4;                      // 0..33
        const int c0 = (idx & 15) * 8;
        v8bf o = {};
        if (w >= 1 && w <= DHW) {
            const float* p = xrow + (size_t)(w - 1) * CIN + c0;
            const float4 f0 = *(const float4*)p;
            const float4 f1 = *(const float4*)(p + 4);
            o[0] = (__bf16)f0.x; o[1] = (__bf16)f0.y;
            o[2] = (__bf16)f0.z; o[3] = (__bf16)f0.w;
            o[4] = (__bf16)f1.x; o[5] = (__bf16)f1.y;
            o[6] = (__bf16)f1.z; o[7] = (__bf16)f1.w;
        }
        *(v8bf*)(orow + (size_t)idx * 8) = o;
    }
}

// =====================================================================
// Kernel 4: implicit-GEMM conv3d via v_wmma_f32_16x16x32_bf16.
// One WG = (b, d, 4 h-rows): M=128 spatial x N=128 filters, K=3456.
// A tile staged into LDS with gfx1250 global_load_async_to_lds_b128.
// =====================================================================
__global__ __launch_bounds__(256)
void conv_wmma_kernel(const __bf16* __restrict__ xw,
                      const __bf16* __restrict__ bpack,
                      const float* __restrict__ demod,
                      const float* __restrict__ b_conv,
                      float* __restrict__ out) {
    extern __shared__ __align__(16) __bf16 lds[];   // [3][6][34][128] bf16

    const int blk  = blockIdx.x;                    // 8*32*8 = 2048
    const int hblk = blk & 7;
    const int d    = (blk >> 3) & 31;
    const int b    = blk >> 8;
    const int h0   = hblk * HROWS;
    const int tid  = threadIdx.x;

    // ---- stage A tile: async copy 18 rows (16B chunks); zero OOB halo ----
    for (int idx = tid; idx < LDS_ROWS * ROW_CHUNKS; idx += 256) {
        const int row   = idx / ROW_CHUNKS;         // 0..17  (dd*6 + hh)
        const int chunk = idx - row * ROW_CHUNKS;   // 0..543
        const int dd = row / 6, hh = row - 6 * (row / 6);
        const int gd = d + dd - 1, gh = h0 + hh - 1;
        __bf16* ldst = lds + (size_t)row * ROWH + (size_t)chunk * 8;
        if (gd >= 0 && gd < DHW && gh >= 0 && gh < DHW) {
            const __bf16* src =
                xw + ((size_t)((b * DHW + gd) * DHW + gh)) * ROWH + (size_t)chunk * 8;
            const unsigned lofs = (unsigned)(unsigned long long)(void*)ldst;
            asm volatile("global_load_async_to_lds_b128 %0, %1, off"
                         :: "v"(lofs), "v"(src) : "memory");
        } else {
            v8bf z = {};
            *(v8bf*)ldst = z;                       // ds_store_b128 of zeros
        }
    }
    asm volatile("s_wait_asynccnt 0" ::: "memory"); // async writes visible
    __syncthreads();

    // ---- GEMM: wave wv owns N-tile wv; 8 M-tile accumulators ----
    const int wv   = tid >> 5;          // 0..7  (N tile)
    const int lane = tid & 31;
    const int l15  = lane & 15;
    const int koff = (lane < 16) ? 0 : 16;  // A-frag K sub-block select

    // loop-invariant per-M-tile LDS half-offsets (A rows: M = lane&15)
    unsigned aofs[MT];
    #pragma unroll
    for (int t = 0; t < MT; ++t) {
        const int m = t * 16 + l15;
        aofs[t] = (unsigned)((((m >> 5) * WPAD) + (m & 31)) * CIN + koff);
    }

    v8f zero = {};
    v8f acc[MT];
    #pragma unroll
    for (int t = 0; t < MT; ++t) acc[t] = zero;

    const __bf16* bp =
        bpack + (((size_t)b * KSTEPS * NT + wv) * 32 + lane) * 16;
    const size_t bstride = (size_t)NT * 32 * 16;    // halfs per k-step

    #pragma unroll 1
    for (int tap = 0; tap < KTAPS; ++tap) {
        const int kw = tap % 3, kh = (tap / 3) % 3, kd = tap / 9;
        const unsigned tapofs = (unsigned)(((kd * 6 + kh) * WPAD + kw) * CIN);
        #pragma unroll
        for (int cblk = 0; cblk < 4; ++cblk) {
            const v16bf bfrag = *(const v16bf*)bp;  // 32B contiguous per lane
            bp += bstride;
            __builtin_prefetch(bp, 0, 1);           // global_prefetch_b8
            const unsigned cb = tapofs + (unsigned)(cblk * 32);
            #pragma unroll
            for (int t = 0; t < MT; ++t) {
                const __bf16* ap = lds + cb + aofs[t];
                // lane<16: K {0..15}; lane>=16: K {16..31}
                const v8bf a0 = *(const v8bf*)ap;           // 16B aligned
                const v8bf a1 = *(const v8bf*)(ap + 8);
                const v16bf afrag = __builtin_shufflevector(
                    a0, a1, 0, 1, 2, 3, 4, 5, 6, 7, 8, 9, 10, 11, 12, 13, 14, 15);
                acc[t] = __builtin_amdgcn_wmma_f32_16x16x32_bf16(
                    false, afrag, false, bfrag, (short)0, acc[t], false, false);
            }
        }
    }

    // ---- epilogue: demod, bias, leaky relu, store f32 NDHWC ----
    const int f    = wv * 16 + l15;     // D cols: N = lane&15
    const float dm = demod[b * CIN + f];
    const float bi = b_conv[f];
    const int mofs = (lane < 16) ? 0 : 8;   // D rows: VGPR r -> M = r (+8 hi lanes)
    #pragma unroll
    for (int t = 0; t < MT; ++t) {
        #pragma unroll
        for (int r = 0; r < 8; ++r) {
            const int m  = t * 16 + mofs + r;
            const int hl = m >> 5, wo = m & 31;
            float v = acc[t][r] * dm + bi;
            v = (v > 0.0f) ? v : 0.2f * v;
            out[((size_t)((b * DHW + d) * DHW + (h0 + hl)) * DHW + wo) * CIN + f] = v;
        }
    }
}

// =====================================================================
// Workspace layout:
//   [0, 4K)            style f32 [8][128]
//   [4K, 8K)           demod f32 [8][128]
//   [8K, 8K+7.08MB)    bpack bf16 [8][108][8][32][16]
//   [+, +71.3MB)       xw bf16 [8][32][32][34][128]
// =====================================================================
#define WS_BPACK_OFF 8192
#define BPACK_HALFS  ((size_t)BATCH * KSTEPS * NT * 32 * 16)      // 3,538,944
#define WS_XW_OFF    (WS_BPACK_OFF + BPACK_HALFS * 2)             // 7,086,080
#define XW_HALFS     ((size_t)BATCH * DHW * DHW * ROWH)           // 35,651,584

extern "C" void kernel_launch(void* const* d_in, const int* in_sizes, int n_in,
                              void* d_out, int out_size, void* d_ws, size_t ws_size,
                              hipStream_t stream) {
    const float* x        = (const float*)d_in[0];
    const float* w_latent = (const float*)d_in[1];
    const float* w_style  = (const float*)d_in[2];
    const float* b_style  = (const float*)d_in[3];
    const float* w_conv   = (const float*)d_in[4];
    const float* b_conv   = (const float*)d_in[5];
    float* out = (float*)d_out;

    char* ws = (char*)d_ws;
    float*  style = (float*)ws;
    float*  dcoef = (float*)(ws + 4096);
    __bf16* bpack = (__bf16*)(ws + WS_BPACK_OFF);
    __bf16* xw    = (__bf16*)(ws + WS_XW_OFF);

    style_demod_kernel<<<BATCH, CIN, 0, stream>>>(w_latent, w_style, b_style,
                                                  w_conv, style, dcoef);
    pack_w_kernel<<<BATCH * KSTEPS * NT, 32, 0, stream>>>(w_conv, style, bpack);
    cvt_pad_kernel<<<BATCH * DHW * DHW, 256, 0, stream>>>(x, xw);

    const size_t smem = (size_t)LDS_ELEMS * sizeof(__bf16);   // 156672 B
    conv_wmma_kernel<<<BATCH * DHW * (DHW / HROWS), 256, smem, stream>>>(
        xw, bpack, dcoef, b_conv, out);
}
